// MaskedSpanBertLikeEncoder_687194768219
// MI455X (gfx1250) — matrix operand order repaired and checked by reading, not compiled, not run on hardware
//
#include <hip/hip_runtime.h>
#include <hip/hip_bf16.h>
#include <math.h>

// MaskedSpanBertLikeEncoder on MI455X (gfx1250), wave32, WMMA f16->f32,
// async global->LDS staging + double-buffered software pipeline.
// B=4 S=512 D=768 I=64 L=2 H=12 F=3072 DH=64

#define BB 4
#define SS 512
#define DD 768
#define II 64
#define LLAY 2
#define HH 12
#define FF 3072
#define DHH 64
#define LN_EPS 1e-12f

typedef _Float16 v16h __attribute__((ext_vector_type(16)));
typedef _Float16 v8h  __attribute__((ext_vector_type(8)));
typedef _Float16 v2h  __attribute__((ext_vector_type(2)));
typedef float    v8f  __attribute__((ext_vector_type(8)));
typedef int      i32x4 __attribute__((vector_size(16)));

#if __has_builtin(__builtin_amdgcn_global_load_async_to_lds_b128)
#define HAVE_ASYNC_LDS 1
#else
#define HAVE_ASYNC_LDS 0
#endif

#define GLOBAL_AS __attribute__((address_space(1)))
#define SHARED_AS __attribute__((address_space(3)))

__device__ __forceinline__ void cp_async_b128(_Float16* ldst, const _Float16* gsrc) {
#if HAVE_ASYNC_LDS
  __builtin_amdgcn_global_load_async_to_lds_b128(
      (GLOBAL_AS i32x4*)gsrc, (SHARED_AS i32x4*)ldst, 0, 0);
#else
  *(v8h*)ldst = *(const v8h*)gsrc;
#endif
}

__device__ __forceinline__ void async_wait() {
#if HAVE_ASYNC_LDS
#if __has_builtin(__builtin_amdgcn_s_wait_asynccnt)
  __builtin_amdgcn_s_wait_asynccnt(0);
#else
  asm volatile("s_wait_asynccnt 0x0" ::: "memory");
#endif
#endif
}

__device__ __forceinline__ v8f wmma16(const v16h a, const v16h b, const v8f c) {
  // v_wmma_f32_16x16x32_f16  D = A*B + C
  return __builtin_amdgcn_wmma_f32_16x16x32_f16(false, a, false, b, (short)0, c, false, false);
}

// 16 contiguous halves starting at p (16B-aligned): two ds_load_b128
__device__ __forceinline__ v16h ld_b16(const _Float16* p) {
  v8h lo = *(const v8h*)p;
  v8h hi = *(const v8h*)(p + 8);
  return __builtin_shufflevector(lo, hi, 0,1,2,3,4,5,6,7,8,9,10,11,12,13,14,15);
}

// A fragment (ISA 16-bit A 16x32 layout): two 8-half runs at kb and kb+16
__device__ __forceinline__ v16h ld_afrag(const _Float16* row, int kb) {
  v8h lo = *(const v8h*)(row + kb);
  v8h hi = *(const v8h*)(row + kb + 16);
  return __builtin_shufflevector(lo, hi, 0,1,2,3,4,5,6,7,8,9,10,11,12,13,14,15);
}

// ---------------------------------------------------------------------------
// 0a) weight pre-pack: W[K][N] f32 -> Wt[N][K] f16 (tiled transpose via LDS)
// grid (N/64, K/64), 256 threads
// ---------------------------------------------------------------------------
__global__ void __launch_bounds__(256) pack_wt_kernel(
    const float* __restrict__ W, _Float16* __restrict__ Wt, int K, int N)
{
  __shared__ __align__(16) _Float16 T[64][72];
  const int tid = threadIdx.x;
  const int n0 = blockIdx.x * 64, k0 = blockIdx.y * 64;
#pragma unroll
  for (int j = 0; j < 16; ++j) {                 // coalesced read along n
    int idx = tid + j * 256; int r = idx >> 6, c = idx & 63;   // r=k, c=n
    T[c][r] = (_Float16)W[(size_t)(k0 + r) * N + n0 + c];
  }
  __syncthreads();
#pragma unroll
  for (int j = 0; j < 2; ++j) {                  // coalesced write along k (v8h)
    int flat = tid + j * 256;                    // 512 x v8h
    int rr = flat >> 3, cg = flat & 7;
    *(v8h*)&Wt[((size_t)(n0 + rr)) * K + k0 + cg * 8] = *(const v8h*)&T[rr][cg * 8];
  }
}

// ---------------------------------------------------------------------------
// 0b) V pre-pack per layer: V[B*S][D] f32 -> Vt[b][d][s] f16
// grid (D/64, S/64, B), 256 threads
// ---------------------------------------------------------------------------
__global__ void __launch_bounds__(256) pack_vt_kernel(
    const float* __restrict__ V, _Float16* __restrict__ Vt)
{
  __shared__ __align__(16) _Float16 T[64][72];
  const int tid = threadIdx.x;
  const int d0 = blockIdx.x * 64, s0 = blockIdx.y * 64, b = blockIdx.z;
#pragma unroll
  for (int j = 0; j < 16; ++j) {                 // coalesced read along d
    int idx = tid + j * 256; int r = idx >> 6, c = idx & 63;   // r=s, c=d
    T[c][r] = (_Float16)V[((size_t)b * SS + s0 + r) * DD + d0 + c];
  }
  __syncthreads();
#pragma unroll
  for (int j = 0; j < 2; ++j) {                  // coalesced write along s (v8h)
    int flat = tid + j * 256;
    int rr = flat >> 3, cg = flat & 7;
    *(v8h*)&Vt[((size_t)b * DD + d0 + rr) * SS + s0 + cg * 8] = *(const v8h*)&T[rr][cg * 8];
  }
}

// ---------------------------------------------------------------------------
// 1) masked max-pool + zero_indic
// ---------------------------------------------------------------------------
__global__ void __launch_bounds__(256) pool_kernel(
    const float* __restrict__ hs0, const unsigned char* __restrict__ mask,
    float* __restrict__ Q, int* __restrict__ zind)
{
  __shared__ unsigned char mrow[SS];
  __shared__ int flag;
  const int tid = threadIdx.x;
  const int bi  = blockIdx.x;            // b*I + i
  const int b   = bi >> 6;
  const unsigned char* mr = mask + (size_t)bi * SS;
  unsigned char m0 = mr[tid], m1 = mr[tid + 256];
  mrow[tid] = m0; mrow[tid + 256] = m1;
  if (tid == 0) flag = 1;
  __syncthreads();
  if (!(m0 && m1)) flag = 0;             // any valid token -> not degenerate
  __syncthreads();
  const int z = flag;                    // 1 == all tokens masked (degenerate span)
  float mv0 = -INFINITY, mv1 = -INFINITY, mv2 = -INFINITY;
  const float* hb = hs0 + (size_t)b * SS * DD;
  for (int s = 0; s < SS; ++s) {
    if (z || !mrow[s]) {                 // degenerate rows get unmasked (ref semantics)
      const float* hr = hb + (size_t)s * DD + tid;
      mv0 = fmaxf(mv0, hr[0]);
      mv1 = fmaxf(mv1, hr[256]);
      mv2 = fmaxf(mv2, hr[512]);
    }
  }
  float* qr = Q + (size_t)bi * DD + tid;
  qr[0] = mv0; qr[256] = mv1; qr[512] = mv2;
  if (tid == 0) zind[bi] = z;
}

// ---------------------------------------------------------------------------
// 2) WMMA GEMM: C[M,N] = A[M,K] @ Wt[N,K]^T + bias, optional exact GELU
// A: f32 row-major; Wt: pre-packed f16 [n][k]
// Double-buffered pipeline: async B-tile copies (ASYNCcnt), A prefetch in regs,
// one barrier per k-step. Block tile 64x128, 8 waves x (2x2) WMMA tiles.
// grid = (N/128, M/64), 256 threads
// ---------------------------------------------------------------------------
__global__ void __launch_bounds__(256) wmma_gemm_kernel(
    const float* __restrict__ A, const _Float16* __restrict__ Wt,
    const float* __restrict__ bias, float* __restrict__ C,
    int M, int N, int K, int epi)
{
  __shared__ __align__(16) _Float16 As[2][64][40];    // 64 x 32 (+pad), x2 buffers
  __shared__ __align__(16) _Float16 BsT[2][128][40];  // [n][k]: 128 x 32 (+pad), x2
  const int tid  = threadIdx.x;
  const int lane = tid & 31, wave = tid >> 5;
  const int wm = wave >> 2, wn = wave & 3;     // 2x4 wave grid
  const int m0 = blockIdx.y * 64;
  const int n0 = blockIdx.x * 128;
  const int kb = (lane >> 4) * 8;              // A frag K sub-offset
  const int rb = (lane >> 4) * 16;             // unused; B frag also uses kb
  const int nsteps = K >> 5;
  v8f acc[2][2];
#pragma unroll
  for (int ti = 0; ti < 2; ++ti)
#pragma unroll
    for (int tj = 0; tj < 2; ++tj)
#pragma unroll
      for (int j = 0; j < 8; ++j) acc[ti][tj][j] = 0.0f;

  // ---- prologue: stage step 0 into buffer 0 ----
#pragma unroll
  for (int j = 0; j < 4; ++j) {                // A tile: 64x32 f32->f16
    int idx = (tid + j * 256) * 2;
    int r = idx >> 5, c = idx & 31;
    const float* p = &A[(size_t)(m0 + r) * K + c];
    v2h pk; pk[0] = (_Float16)p[0]; pk[1] = (_Float16)p[1];
    *(v2h*)&As[0][r][c] = pk;
  }
#pragma unroll
  for (int j = 0; j < 2; ++j) {                // B tile: async global->LDS b128
    int flat = tid + j * 256;
    int n = flat >> 2, kg = flat & 3;
    cp_async_b128(&BsT[0][n][kg * 8], &Wt[(size_t)(n0 + n) * K + kg * 8]);
  }
  async_wait();
  __syncthreads();

  for (int i = 0; i < nsteps; ++i) {
    const int cur = i & 1, nxt = cur ^ 1;
    const int k1 = (i + 1) << 5;
    const bool more = (i + 1) < nsteps;        // uniform
    float pf[4][2];
    if (more) {
#pragma unroll
      for (int j = 0; j < 4; ++j) {            // prefetch next A into regs
        int idx = (tid + j * 256) * 2;
        int r = idx >> 5, c = idx & 31;
        const float* p = &A[(size_t)(m0 + r) * K + k1 + c];
        pf[j][0] = p[0]; pf[j][1] = p[1];
      }
#pragma unroll
      for (int j = 0; j < 2; ++j) {            // fire async copy of next B
        int flat = tid + j * 256;
        int n = flat >> 2, kg = flat & 3;
        cp_async_b128(&BsT[nxt][n][kg * 8], &Wt[(size_t)(n0 + n) * K + k1 + kg * 8]);
      }
    }
    // ---- compute current step ----
    v16h af[2], bf[2];
#pragma unroll
    for (int ti = 0; ti < 2; ++ti)
      af[ti] = ld_afrag(&As[cur][wm * 32 + ti * 16 + (lane & 15)][0], kb);
#pragma unroll
    for (int tj = 0; tj < 2; ++tj)
      bf[tj] = ld_afrag(&BsT[cur][wn * 32 + tj * 16 + (lane & 15)][0], kb);
#pragma unroll
    for (int ti = 0; ti < 2; ++ti)
#pragma unroll
      for (int tj = 0; tj < 2; ++tj)
        acc[ti][tj] = wmma16(af[ti], bf[tj], acc[ti][tj]);
    // ---- commit next A tile (convert + LDS store) ----
    if (more) {
#pragma unroll
      for (int j = 0; j < 4; ++j) {
        int idx = (tid + j * 256) * 2;
        int r = idx >> 5, c = idx & 31;
        v2h pk; pk[0] = (_Float16)pf[j][0]; pk[1] = (_Float16)pf[j][1];
        *(v2h*)&As[nxt][r][c] = pk;
      }
    }
    async_wait();
    __syncthreads();
  }

#pragma unroll
  for (int ti = 0; ti < 2; ++ti)
#pragma unroll
    for (int tj = 0; tj < 2; ++tj) {
      int col = n0 + wn * 32 + tj * 16 + (lane & 15);
      float bv = bias[col];
#pragma unroll
      for (int j = 0; j < 8; ++j) {
        int row = m0 + wm * 32 + ti * 16 + (lane >> 4) * 8 + j;
        float v = acc[ti][tj][j] + bv;
        if (epi == 1) v = 0.5f * v * (1.0f + erff(v * 0.70710678118654752f)); // exact GELU
        C[(size_t)row * N + col] = v;
      }
    }
}

// ---------------------------------------------------------------------------
// 3) attention per (b,h): grid = B*H, 256 threads
//    scores (64x512, f32, LDS) = (Q/8) @ K^T + mask_bias; softmax; ctx = P @ V
//    K chunks natural [s][d]; V pre-packed f16 [b][d][s] staged via async copy.
// ---------------------------------------------------------------------------
__global__ void __launch_bounds__(256) attn_kernel(
    const float* __restrict__ Qh, const float* __restrict__ Kh,
    const _Float16* __restrict__ Vt, const unsigned char* __restrict__ mask,
    const int* __restrict__ zind, float* __restrict__ Ctx)
{
  __shared__ __align__(16) _Float16 Qs[II][DHH + 8];
  __shared__ __align__(16) _Float16 Ts[64][DHH + 8]; // K chunk [s][d]
  __shared__ __align__(16) _Float16 Vs[64][40];      // V chunk [d][s] (pre-packed)
  __shared__ __align__(32) float    Scm[II][SS];     // 128 KB (CDNA5: 320KB LDS/WGP)
  const int tid  = threadIdx.x;
  const int lane = tid & 31, wave = tid >> 5;
  const int wm = wave >> 2, wn = wave & 3;
  const int kb = (lane >> 4) * 8;
  const int rb = (lane >> 4) * 16;
  const int b = blockIdx.x / HH, h = blockIdx.x % HH;
  const float* Qb = Qh + (size_t)b * II * DD + h * DHH;
  const float* Kb = Kh + (size_t)b * SS * DD + h * DHH;
  const _Float16* Vb = Vt + ((size_t)b * DD + h * DHH) * SS;

  // load Q, pre-scaled by 1/sqrt(DH) = 0.125; 2 consecutive halves per thread
#pragma unroll
  for (int j = 0; j < 8; ++j) {
    int idx = (tid + j * 256) * 2; int r = idx >> 6, c = idx & 63;
    const float* p = &Qb[(size_t)r * DD + c];
    v2h pk; pk[0] = (_Float16)(p[0] * 0.125f); pk[1] = (_Float16)(p[1] * 0.125f);
    *(v2h*)&Qs[r][c] = pk;
  }
  __syncthreads();

  // ---- scores: 8 chunks of 64 keys ----
  for (int sc = 0; sc < 8; ++sc) {
#pragma unroll
    for (int j = 0; j < 8; ++j) {        // K chunk natural [s_local][d]
      int idx = (tid + j * 256) * 2; int r = idx >> 6, c = idx & 63;
      const float* p = &Kb[(size_t)(sc * 64 + r) * DD + c];
      v2h pk; pk[0] = (_Float16)p[0]; pk[1] = (_Float16)p[1];
      *(v2h*)&Ts[r][c] = pk;
    }
    __syncthreads();
    v8f acc[2];
#pragma unroll
    for (int ti = 0; ti < 2; ++ti)
#pragma unroll
      for (int j = 0; j < 8; ++j) acc[ti][j] = 0.0f;
#pragma unroll
    for (int k0 = 0; k0 < DHH; k0 += 32) {
      v16h af[2], bfv;
#pragma unroll
      for (int ti = 0; ti < 2; ++ti)
        af[ti] = ld_afrag(&Qs[wm * 32 + ti * 16 + (lane & 15)][k0], kb);
      bfv = ld_afrag(&Ts[wn * 16 + (lane & 15)][k0], kb);
#pragma unroll
      for (int ti = 0; ti < 2; ++ti) acc[ti] = wmma16(af[ti], bfv, acc[ti]);
    }
#pragma unroll
    for (int ti = 0; ti < 2; ++ti)
#pragma unroll
      for (int j = 0; j < 8; ++j) {
        int i  = wm * 32 + ti * 16 + (lane >> 4) * 8 + j;
        int sl = wn * 16 + (lane & 15);
        int s  = sc * 64 + sl;
        int z  = zind[b * II + i];
        float e = (!z && mask[((size_t)b * II + i) * SS + s]) ? -10000.0f : 0.0f;
        Scm[i][s] = acc[ti][j] + e;
      }
    __syncthreads();
  }

  // ---- softmax over S=512, one wave owns 8 rows ----
  for (int rr = 0; rr < 8; ++rr) {
    int r = wave * 8 + rr;
    float vals[16]; float m = -INFINITY;
#pragma unroll
    for (int k = 0; k < 16; ++k) { vals[k] = Scm[r][lane + k * 32]; m = fmaxf(m, vals[k]); }
#pragma unroll
    for (int off = 16; off; off >>= 1) m = fmaxf(m, __shfl_xor(m, off, 32));
    float sum = 0.0f;
#pragma unroll
    for (int k = 0; k < 16; ++k) { vals[k] = expf(vals[k] - m); sum += vals[k]; }
#pragma unroll
    for (int off = 16; off; off >>= 1) sum += __shfl_xor(sum, off, 32);
    float inv = 1.0f / sum;
#pragma unroll
    for (int k = 0; k < 16; ++k) Scm[r][lane + k * 32] = vals[k] * inv;
  }
  __syncthreads();

  // ---- ctx = P(64x512) @ V(512x64), 16 K-chunks of 32 ----
  v8f cacc[2];
#pragma unroll
  for (int ti = 0; ti < 2; ++ti)
#pragma unroll
    for (int j = 0; j < 8; ++j) cacc[ti][j] = 0.0f;
  for (int sc2 = 0; sc2 < 16; ++sc2) {
    {                                    // V chunk [d][32s]: one async b128 per thread
      int d = tid >> 2, kg = tid & 3;
      cp_async_b128(&Vs[d][kg * 8], &Vb[(size_t)d * SS + sc2 * 32 + kg * 8]);
    }
    async_wait();
    __syncthreads();
    v16h bfv = ld_afrag(&Vs[wn * 16 + (lane & 15)][0], kb);
#pragma unroll
    for (int ti = 0; ti < 2; ++ti) {
      int r = wm * 32 + ti * 16 + (lane & 15);
      const float* pr = &Scm[r][sc2 * 32 + kb];
      v8f lo = *(const v8f*)pr;          // K kb..kb+7
      v8f hi = *(const v8f*)(pr + 16);   // K kb+16..kb+23
      v16h af;
#pragma unroll
      for (int e = 0; e < 8; ++e) { af[e] = (_Float16)lo[e]; af[e + 8] = (_Float16)hi[e]; }
      cacc[ti] = wmma16(af, bfv, cacc[ti]);
    }
    __syncthreads();
  }
#pragma unroll
  for (int ti = 0; ti < 2; ++ti)
#pragma unroll
    for (int j = 0; j < 8; ++j) {
      int i = wm * 32 + ti * 16 + (lane >> 4) * 8 + j;
      int d = wn * 16 + (lane & 15);
      Ctx[((size_t)b * II + i) * DD + h * DHH + d] = cacc[ti][j];
    }
}

// ---------------------------------------------------------------------------
// 4) residual + LayerNorm: Out = LN(X+Y)*g + b   (in-place safe, Out may == X)
// ---------------------------------------------------------------------------
__global__ void __launch_bounds__(256) add_ln_kernel(
    const float* __restrict__ X, const float* __restrict__ Y,
    const float* __restrict__ g, const float* __restrict__ bta,
    float* __restrict__ Out)
{
  __shared__ float red[256];
  const int tid = threadIdx.x;
  const size_t row = (size_t)blockIdx.x * DD;
  float s[3];
#pragma unroll
  for (int j = 0; j < 3; ++j) { int c = tid + j * 256; s[j] = X[row + c] + Y[row + c]; }
  float loc = s[0] + s[1] + s[2];
  red[tid] = loc; __syncthreads();
  for (int off = 128; off > 0; off >>= 1) { if (tid < off) red[tid] += red[tid + off]; __syncthreads(); }
  float mean = red[0] * (1.0f / DD);
  __syncthreads();
  loc = 0.0f;
#pragma unroll
  for (int j = 0; j < 3; ++j) { float d = s[j] - mean; loc += d * d; }
  red[tid] = loc; __syncthreads();
  for (int off = 128; off > 0; off >>= 1) { if (tid < off) red[tid] += red[tid + off]; __syncthreads(); }
  float rs = rsqrtf(red[0] * (1.0f / DD) + LN_EPS);
#pragma unroll
  for (int j = 0; j < 3; ++j) {
    int c = tid + j * 256;
    Out[row + c] = (s[j] - mean) * rs * g[c] + bta[c];
  }
}

// ---------------------------------------------------------------------------
// 5) zero-context substitution
// ---------------------------------------------------------------------------
__global__ void __launch_bounds__(256) final_kernel(
    const float* __restrict__ Q, const int* __restrict__ zind,
    const float* __restrict__ zc, float* __restrict__ out)
{
  const int tid = threadIdx.x;
  const int bi = blockIdx.x;
  const int z = zind[bi];
  const size_t row = (size_t)bi * DD;
#pragma unroll
  for (int j = 0; j < 3; ++j) {
    int c = tid + j * 256;
    out[row + c] = z ? zc[c] : Q[row + c];
  }
}

// ---------------------------------------------------------------------------
extern "C" void kernel_launch(void* const* d_in, const int* in_sizes, int n_in,
                              void* d_out, int out_size, void* d_ws, size_t ws_size,
                              hipStream_t stream) {
  const float* hs0  = (const float*)d_in[0];
  const float* hs1  = (const float*)d_in[1];
  const float* Wq   = (const float*)d_in[3];
  const float* bq   = (const float*)d_in[4];
  const float* Wk   = (const float*)d_in[5];
  const float* bk   = (const float*)d_in[6];
  const float* Wv   = (const float*)d_in[7];
  const float* bv   = (const float*)d_in[8];
  const float* Wo   = (const float*)d_in[9];
  const float* bo   = (const float*)d_in[10];
  const float* ln1g = (const float*)d_in[11];
  const float* ln1b = (const float*)d_in[12];
  const float* W1   = (const float*)d_in[13];
  const float* b1   = (const float*)d_in[14];
  const float* W2   = (const float*)d_in[15];
  const float* b2   = (const float*)d_in[16];
  const float* ln2g = (const float*)d_in[17];
  const float* ln2b = (const float*)d_in[18];
  const float* zc   = (const float*)d_in[19];
  const unsigned char* mask = (const unsigned char*)d_in[20];

  // ---- workspace layout ----
  float* ws    = (float*)d_ws;
  float* wsQ   = ws;                              // [B*I, D]
  float* wsK   = wsQ   + (size_t)BB*II*DD;        // [B*S, D]
  float* wsV   = wsK   + (size_t)BB*SS*DD;        // [B*S, D]
  float* wsQh  = wsV   + (size_t)BB*SS*DD;        // [B*I, D]
  float* wsCtx = wsQh  + (size_t)BB*II*DD;        // [B*I, D]
  float* wsT   = wsCtx + (size_t)BB*II*DD;        // [B*I, D]
  float* wsF   = wsT   + (size_t)BB*II*DD;        // [B*I, F]
  int*   zind  = (int*)(wsF + (size_t)BB*II*FF);  // [B*I]
  // f16 region (16B aligned: all prior sizes are multiples of 16B)
  _Float16* wtBase = (_Float16*)(zind + 256);
  const size_t SZ_DD = (size_t)DD * DD, SZ_DF = (size_t)DD * FF;
  const size_t perLayer = 4 * SZ_DD + 2 * SZ_DF;
  _Float16* wtQ = wtBase;                  // per layer: Q,K,V,O [D*D] then W1 [D*F], W2 [F*D]
  _Float16* wsVt = wtQ + LLAY * perLayer;          // [B, D, S] f16

  // ---- one-time weight pre-pack (transpose + f16) ----
  const dim3 gPackDD(DD / 64, DD / 64);    // (12,12)
  const dim3 gPackDF(FF / 64, DD / 64);    // W1: K=DD,N=FF -> (48,12)
  const dim3 gPackFD(DD / 64, FF / 64);    // W2: K=FF,N=DD -> (12,48)
  for (int l = 0; l < LLAY; ++l) {
    _Float16* base = wtQ + (size_t)l * perLayer;
    pack_wt_kernel<<<gPackDD, 256, 0, stream>>>(Wq + (size_t)l*SZ_DD, base,             DD, DD);
    pack_wt_kernel<<<gPackDD, 256, 0, stream>>>(Wk + (size_t)l*SZ_DD, base + SZ_DD,     DD, DD);
    pack_wt_kernel<<<gPackDD, 256, 0, stream>>>(Wv + (size_t)l*SZ_DD, base + 2*SZ_DD,   DD, DD);
    pack_wt_kernel<<<gPackDD, 256, 0, stream>>>(Wo + (size_t)l*SZ_DD, base + 3*SZ_DD,   DD, DD);
    pack_wt_kernel<<<gPackDF, 256, 0, stream>>>(W1 + (size_t)l*SZ_DF, base + 4*SZ_DD,   DD, FF);
    pack_wt_kernel<<<gPackFD, 256, 0, stream>>>(W2 + (size_t)l*SZ_DF, base + 4*SZ_DD + SZ_DF, FF, DD);
  }

  pool_kernel<<<BB*II, 256, 0, stream>>>(hs0, mask, wsQ, zind);

  const dim3 gKV (DD / 128, (BB * SS) / 64);   // (6, 32)
  const dim3 gQ  (DD / 128, (BB * II) / 64);   // (6, 4)
  const dim3 gF1 (FF / 128, (BB * II) / 64);   // (24, 4)
  const dim3 gVt (DD / 64, SS / 64, BB);       // (12, 8, 4)

  for (int l = 0; l < LLAY; ++l) {
    const float* hs = (l == 0) ? hs0 : hs1;
    _Float16* base = wtQ + (size_t)l * perLayer;
    const _Float16 *tWq = base, *tWk = base + SZ_DD, *tWv = base + 2*SZ_DD,
                   *tWo = base + 3*SZ_DD, *tW1 = base + 4*SZ_DD, *tW2 = base + 4*SZ_DD + SZ_DF;
    wmma_gemm_kernel<<<gKV, 256, 0, stream>>>(hs,   tWk, bk + l*DD, wsK, BB*SS, DD, DD, 0);
    wmma_gemm_kernel<<<gKV, 256, 0, stream>>>(hs,   tWv, bv + l*DD, wsV, BB*SS, DD, DD, 0);
    pack_vt_kernel<<<gVt, 256, 0, stream>>>(wsV, wsVt);
    wmma_gemm_kernel<<<gQ,  256, 0, stream>>>(wsQ,  tWq, bq + l*DD, wsQh, BB*II, DD, DD, 0);
    attn_kernel<<<BB*HH, 256, 0, stream>>>(wsQh, wsK, wsVt, mask, zind, wsCtx);
    wmma_gemm_kernel<<<gQ,  256, 0, stream>>>(wsCtx, tWo, bo + l*DD, wsT, BB*II, DD, DD, 0);
    add_ln_kernel<<<BB*II, 256, 0, stream>>>(wsQ, wsT, ln1g + l*DD, ln1b + l*DD, wsQ);
    wmma_gemm_kernel<<<gF1, 256, 0, stream>>>(wsQ, tW1, b1 + l*FF, wsF, BB*II, FF, DD, 1);
    wmma_gemm_kernel<<<gQ,  256, 0, stream>>>(wsF, tW2, b2 + l*DD, wsT, BB*II, DD, FF, 0);
    add_ln_kernel<<<BB*II, 256, 0, stream>>>(wsQ, wsT, ln2g + l*DD, ln2b + l*DD, wsQ);
  }

  final_kernel<<<BB*II, 256, 0, stream>>>(wsQ, zind, zc, (float*)d_out);
}
